// DecoupleBranch_87454124081651
// MI455X (gfx1250) — matrix operand order repaired and checked by reading, not compiled, not run on hardware
//
#include <hip/hip_runtime.h>
#include <hip/hip_bf16.h>

// ---------------------------------------------------------------------------
// CDNA5 (gfx1250) deformable-conv decoupled branch.
// bf16 WMMA (v_wmma_f32_16x16x32_bf16) for the 1x1 conv and the deform GEMM,
// fp32 accumulation everywhere, LDS staging in WMMA fragment layout.
// ---------------------------------------------------------------------------

typedef __attribute__((ext_vector_type(16))) __bf16 v16bf;
typedef __attribute__((ext_vector_type(8)))  float  v8f;

union FragBF {
    uint4 u[2];
    v16bf v;
};

__device__ __forceinline__ unsigned short f2bf_u(float f) {
    union { float f; unsigned u; } c; c.f = f;
    unsigned r = c.u + 0x7FFFu + ((c.u >> 16) & 1u);   // round-to-nearest-even
    return (unsigned short)(r >> 16);
}
__device__ __forceinline__ float bf2f(unsigned short u) {
    union { unsigned u; float f; } c; c.u = ((unsigned)u) << 16;
    return c.f;
}

// ---------------------------------------------------------------------------
// Kernel 0: weight preparation.
//   dwbf[o*256+i]            = bf16(dw[o,i])                       (1x1 conv A)
//   twbf[o*2304 + t*256+ci]  = bf16(tw[o,ci,t])                    (deform A, K=(tap,ci))
//   owr [(o*9+t)*256 + ci]   = ow[o,ci,t]   (fp32, channel-last)   (offset conv W)
// ---------------------------------------------------------------------------
__global__ void prep_weights(const float* __restrict__ dw,
                             const float* __restrict__ tw,
                             const float* __restrict__ ow,
                             unsigned short* __restrict__ dwbf,
                             unsigned short* __restrict__ twbf,
                             float* __restrict__ owr)
{
    const int NT_DW = 256 * 256;
    const int NT_TW = 256 * 2304;
    const int NT_OW = 18 * 2304;
    const int total = NT_DW + NT_TW + NT_OW;
    for (int i = blockIdx.x * blockDim.x + threadIdx.x; i < total;
         i += gridDim.x * blockDim.x) {
        if (i < NT_DW) {
            dwbf[i] = f2bf_u(dw[i]);
        } else if (i < NT_DW + NT_TW) {
            int j = i - NT_DW;
            int o = j / 2304, r = j % 2304, t = r >> 8, ci = r & 255;
            twbf[j] = f2bf_u(tw[(o * 256 + ci) * 9 + t]);
        } else {
            int j = i - NT_DW - NT_TW;
            int o = j / 2304, r = j % 2304, t = r >> 8, ci = r & 255;
            owr[j] = ow[(o * 256 + ci) * 9 + t];
        }
    }
}

// ---------------------------------------------------------------------------
// Kernel 1: 1x1 conv as WMMA GEMM.  M=Co=256, K=Ci=256, N=16 pixels / block.
// Output h stored bf16 channel-last: h[pix][256].
// B fragment LDS layout per K-block kb (1 KB): lane = 16*hi + n, bytes j*2,
// element j holds K = kb*32 + 16*hi + j, column n.
// ---------------------------------------------------------------------------
__global__ __launch_bounds__(256)
void conv1x1_wmma(const float* __restrict__ feat,
                  const unsigned short* __restrict__ dwbf,
                  const float* __restrict__ bias,
                  unsigned short* __restrict__ h)
{
    __shared__ __align__(16) unsigned char Bpanel[8 * 1024];  // 8 K-blocks x 1KB

    const int tid  = threadIdx.x;
    const int pix0 = blockIdx.x * 16;           // 16 consecutive x, same b,y
    const int b    = pix0 >> 12;
    const int p    = pix0 & 4095;               // y*64 + x0 within image

    // ---- Phase A: stage features tile (256 ci x 16 px) to LDS as B fragments
#pragma unroll
    for (int rep = 0; rep < 16; ++rep) {
        int idx = rep * 256 + tid;              // 4096 elements
        int ci = idx >> 4, n = idx & 15;        // 16 consecutive threads = 16 x's
        float v = feat[(b * 256 + ci) * 4096 + p + n];
        int kb = ci >> 5, kk = ci & 31;
        int lane = ((kk >> 4) << 4) + n;
        *(unsigned short*)(Bpanel + kb * 1024 + lane * 32 + (kk & 15) * 2) = f2bf_u(v);
    }
    __syncthreads();

    // ---- Phase B: 8 waves, each owns two 16x16 M-tiles
    const int w = tid >> 5, lane = tid & 31;
    const int hi = lane >> 4, lm = lane & 15;
    const int m0 = w * 16, m1 = (w + 8) * 16;
    v8f acc0 = {}, acc1 = {};

#pragma unroll
    for (int kb = 0; kb < 8; ++kb) {
        FragBF fb;
        fb.u[0] = *(const uint4*)(Bpanel + kb * 1024 + lane * 32);
        fb.u[1] = *(const uint4*)(Bpanel + kb * 1024 + lane * 32 + 16);
        const unsigned short* a0p = dwbf + (m0 + lm) * 256 + kb * 32 + 16 * hi;
        const unsigned short* a1p = dwbf + (m1 + lm) * 256 + kb * 32 + 16 * hi;
        FragBF fa0, fa1;
        fa0.u[0] = *(const uint4*)a0p; fa0.u[1] = *(const uint4*)(a0p + 8);
        fa1.u[0] = *(const uint4*)a1p; fa1.u[1] = *(const uint4*)(a1p + 8);
        acc0 = __builtin_amdgcn_wmma_f32_16x16x32_bf16(false, fa0.v, false, fb.v,
                                                       (short)0, acc0, false, false);
        acc1 = __builtin_amdgcn_wmma_f32_16x16x32_bf16(false, fa1.v, false, fb.v,
                                                       (short)0, acc1, false, false);
    }

    // ---- Epilogue: bias, pack 8 consecutive channels to bf16, 16B store
    const int pix = pix0 + lm;
    {
        int cb = m0 + 8 * hi;
        uint4 pk; unsigned* pu = (unsigned*)&pk;
#pragma unroll
        for (int r = 0; r < 8; r += 2) {
            float v0 = acc0[r]     + bias[cb + r];
            float v1 = acc0[r + 1] + bias[cb + r + 1];
            pu[r >> 1] = (unsigned)f2bf_u(v0) | ((unsigned)f2bf_u(v1) << 16);
        }
        *(uint4*)(h + (size_t)pix * 256 + cb) = pk;
    }
    {
        int cb = m1 + 8 * hi;
        uint4 pk; unsigned* pu = (unsigned*)&pk;
#pragma unroll
        for (int r = 0; r < 8; r += 2) {
            float v0 = acc1[r]     + bias[cb + r];
            float v1 = acc1[r + 1] + bias[cb + r + 1];
            pu[r >> 1] = (unsigned)f2bf_u(v0) | ((unsigned)f2bf_u(v1) << 16);
        }
        *(uint4*)(h + (size_t)pix * 256 + cb) = pk;
    }
}

// ---------------------------------------------------------------------------
// Kernel 2: 3x3 offset-predicting conv (Co=18), fp32 VALU.
// block = 288 threads = 16 pixels x 18 output channels.
// offs[pix][18] fp32 (channel 2t = dy, 2t+1 = dx for tap t).
// ---------------------------------------------------------------------------
__global__ __launch_bounds__(288)
void offset_conv(const unsigned short* __restrict__ h,
                 const float* __restrict__ owr,
                 const float* __restrict__ ob,
                 float* __restrict__ offs)
{
    const int tid = threadIdx.x;
    const int pl = tid / 18, o = tid % 18;
    const int pix = blockIdx.x * 16 + pl;
    const int b = pix >> 12, y = (pix >> 6) & 63, x = pix & 63;

    float acc = ob[o];
#pragma unroll
    for (int t = 0; t < 9; ++t) {
        int yy = y + t / 3 - 1;
        int xx = x + t % 3 - 1;
        if (yy < 0 || yy > 63 || xx < 0 || xx > 63) continue;
        const unsigned short* hp = h + ((size_t)(b * 4096 + yy * 64 + xx)) * 256;
        const float* wp = owr + (o * 9 + t) * 256;
        for (int c = 0; c < 256; c += 8) {
            uint4 u = *(const uint4*)(hp + c);
            const unsigned short* us = (const unsigned short*)&u;
            float4 wa = *(const float4*)(wp + c);
            float4 wb = *(const float4*)(wp + c + 4);
            acc += bf2f(us[0]) * wa.x + bf2f(us[1]) * wa.y +
                   bf2f(us[2]) * wa.z + bf2f(us[3]) * wa.w +
                   bf2f(us[4]) * wb.x + bf2f(us[5]) * wb.y +
                   bf2f(us[6]) * wb.z + bf2f(us[7]) * wb.w;
        }
    }
    offs[(size_t)pix * 18 + o] = acc;
}

// ---------------------------------------------------------------------------
// Kernel 3: deformable 3x3 conv as WMMA GEMM.  M=Co=256, N=16 px, K=2304.
// 3 phases of 3 taps (K=768 each): fill B panel (bilinear gather -> LDS in
// fragment layout, 24KB), then 24 WMMAs per wave per phase.
// ---------------------------------------------------------------------------
__global__ __launch_bounds__(512)
void deform_wmma(const unsigned short* __restrict__ h,
                 const float* __restrict__ offs,
                 const unsigned short* __restrict__ twbf,
                 const float* __restrict__ bias,
                 float* __restrict__ out)
{
    __shared__ __align__(16) unsigned char Bpanel[24 * 1024];  // 768 K x 16 N bf16

    const int tid  = threadIdx.x;
    const int w    = tid >> 5, lane = tid & 31;
    const int hi   = lane >> 4, lm = lane & 15;
    const int pix0 = blockIdx.x * 16;
    const int b    = pix0 >> 12;
    const int y    = (pix0 >> 6) & 63;
    const int x0   = pix0 & 63;
    const int M0   = w * 16;

    v8f acc = {};

    for (int tg = 0; tg < 3; ++tg) {
        // ---------------- fill: 48 (pixel,tap) pairs, one wave per pair ----
        if (tg) __syncthreads();        // protect previous compute phase reads
#pragma unroll
        for (int rnd = 0; rnd < 3; ++rnd) {
            int pp   = rnd * 16 + w;    // 0..47
            int n    = pp / 3;
            int tloc = pp % 3;
            int t    = tg * 3 + tloc;
            int pix  = pix0 + n;
            int x    = x0 + n;

            float dy = offs[(size_t)pix * 18 + 2 * t];
            float dx = offs[(size_t)pix * 18 + 2 * t + 1];
            float sy = (float)(y + t / 3 - 1) + dy;
            float sx = (float)(x + t % 3 - 1) + dx;
            float fy = floorf(sy), fx = floorf(sx);
            float ty = sy - fy,    tx = sx - fx;
            int iy0 = (int)fy, ix0 = (int)fx;
            float w00 = (1.f - ty) * (1.f - tx);
            float w01 = (1.f - ty) * tx;
            float w10 = ty * (1.f - tx);
            float w11 = ty * tx;

            const int c0 = lane * 8;    // 8 channels per lane
            float s[8];
#pragma unroll
            for (int i = 0; i < 8; ++i) s[i] = 0.f;

#pragma unroll
            for (int corner = 0; corner < 4; ++corner) {
                int iy = iy0 + (corner >> 1);
                int ix = ix0 + (corner & 1);
                float wt = (corner == 0) ? w00 : (corner == 1) ? w01
                         : (corner == 2) ? w10 : w11;
                if (iy >= 0 && iy < 64 && ix >= 0 && ix < 64) {
                    const unsigned short* hp =
                        h + ((size_t)(b * 4096 + iy * 64 + ix)) * 256 + c0;
                    uint4 u = *(const uint4*)hp;
                    const unsigned short* us = (const unsigned short*)&u;
#pragma unroll
                    for (int i = 0; i < 8; ++i) s[i] += wt * bf2f(us[i]);
                }
            }

            // pack 8 bf16, store into fragment layout
            uint4 pk; unsigned* pu = (unsigned*)&pk;
#pragma unroll
            for (int i = 0; i < 8; i += 2)
                pu[i >> 1] = (unsigned)f2bf_u(s[i]) | ((unsigned)f2bf_u(s[i + 1]) << 16);

            int kl = tloc * 256 + c0;           // local K within this phase
            int kb = kl >> 5, kk = kl & 31;     // kk in {0,8,16,24}
            unsigned char* dst = Bpanel + kb * 1024 +
                                 (((kk >> 4) << 4) + n) * 32 + ((kk & 15) >= 8 ? 16 : 0);
            *(uint4*)dst = pk;
        }
        __syncthreads();

        // ---------------- compute: 24 WMMAs per wave --------------------
        const int koff0 = tg * 768;
        const unsigned short* arow = twbf + (size_t)(M0 + lm) * 2304 + koff0 + 16 * hi;
#pragma unroll 4
        for (int kbl = 0; kbl < 24; ++kbl) {
            FragBF fb;
            fb.u[0] = *(const uint4*)(Bpanel + kbl * 1024 + lane * 32);
            fb.u[1] = *(const uint4*)(Bpanel + kbl * 1024 + lane * 32 + 16);
            const unsigned short* ap = arow + kbl * 32;
            FragBF fa;
            fa.u[0] = *(const uint4*)ap;
            fa.u[1] = *(const uint4*)(ap + 8);
            __builtin_prefetch(ap + 64, 0, 0);   // global_prefetch for next A chunk
            acc = __builtin_amdgcn_wmma_f32_16x16x32_bf16(false, fa.v, false, fb.v,
                                                          (short)0, acc, false, false);
        }
    }

    // ---------------- epilogue: bias + relu, NCHW fp32 store ------------
    const int x = x0 + lm;
#pragma unroll
    for (int r = 0; r < 8; ++r) {
        int m = M0 + 8 * hi + r;
        float v = acc[r] + bias[m];
        v = v > 0.f ? v : 0.f;
        out[((size_t)(b * 256 + m) * 64 + y) * 64 + x] = v;
    }
}

// ---------------------------------------------------------------------------
// Host-side launch
// ---------------------------------------------------------------------------
extern "C" void kernel_launch(void* const* d_in, const int* in_sizes, int n_in,
                              void* d_out, int out_size, void* d_ws, size_t ws_size,
                              hipStream_t stream) {
    (void)in_sizes; (void)n_in; (void)out_size; (void)ws_size;

    const float* feat  = (const float*)d_in[0];
    const float* dw[2] = {(const float*)d_in[1], (const float*)d_in[7]};
    const float* db[2] = {(const float*)d_in[2], (const float*)d_in[8]};
    const float* ow[2] = {(const float*)d_in[3], (const float*)d_in[9]};
    const float* ob[2] = {(const float*)d_in[4], (const float*)d_in[10]};
    const float* tw[2] = {(const float*)d_in[5], (const float*)d_in[11]};
    const float* tb[2] = {(const float*)d_in[6], (const float*)d_in[12]};
    float* out = (float*)d_out;

    char* ws = (char*)d_ws;
    size_t off = 0;
    auto alloc = [&](size_t bytes) -> char* {
        char* p = ws + off;
        off += (bytes + 255) & ~(size_t)255;
        return p;
    };
    unsigned short* dwbf[2]; unsigned short* twbf[2]; float* owr[2];
    unsigned short* hbuf[2]; float* offsb[2];
    for (int br = 0; br < 2; ++br) dwbf[br]  = (unsigned short*)alloc(256 * 256 * 2);
    for (int br = 0; br < 2; ++br) twbf[br]  = (unsigned short*)alloc(256 * 2304 * 2);
    for (int br = 0; br < 2; ++br) owr[br]   = (float*)alloc(18 * 2304 * 4);
    for (int br = 0; br < 2; ++br) hbuf[br]  = (unsigned short*)alloc((size_t)16384 * 256 * 2);
    for (int br = 0; br < 2; ++br) offsb[br] = (float*)alloc((size_t)16384 * 18 * 4);

    for (int br = 0; br < 2; ++br)
        prep_weights<<<512, 256, 0, stream>>>(dw[br], tw[br], ow[br],
                                              dwbf[br], twbf[br], owr[br]);
    for (int br = 0; br < 2; ++br)
        conv1x1_wmma<<<1024, 256, 0, stream>>>(feat, dwbf[br], db[br], hbuf[br]);
    for (int br = 0; br < 2; ++br)
        offset_conv<<<1024, 288, 0, stream>>>(hbuf[br], owr[br], ob[br], offsb[br]);
    for (int br = 0; br < 2; ++br)
        deform_wmma<<<1024, 512, 0, stream>>>(hbuf[br], offsb[br], twbf[br], tb[br],
                                              out + (size_t)br * 4194304);
}
